// DualStreamMamba_48481590837737
// MI455X (gfx1250) — matrix operand order repaired and checked by reading, not compiled, not run on hardware
//
#include <hip/hip_runtime.h>
#include <hip/hip_bf16.h>
#include <math.h>

// ---------------------------------------------------------------------------
// DualStreamMamba on gfx1250 (MI455X).
// All big einsums -> one generic NT WMMA GEMM (C[M,N] = A[M,K] * B[N,K]^T),
// f32 memory, f16 WMMA compute with f32 accumulation.
// Block tile 128x64, 8 wave32s, each wave computes a 32x32 patch via
// 2x2 register-blocked 16x16x32 WMMAs (4 WMMA per wave per K-step).
// ---------------------------------------------------------------------------

typedef _Float16 v16h __attribute__((ext_vector_type(16)));
typedef _Float16 v8h  __attribute__((ext_vector_type(8)));
typedef float    v8f  __attribute__((ext_vector_type(8)));
typedef float    v4f  __attribute__((ext_vector_type(4)));

#define TM 128     // block tile M
#define TN 64      // block tile N
#define KT 32      // K step (matches 16x16x32 f16 WMMA)
#define AP 40      // LDS pitch (halves): 80B rows -> 16B-aligned fragments
#define BP 40

__device__ __forceinline__ v8h cvt_v8(const float* __restrict__ p) {
    v4f a0 = *(const v4f*)p;
    v4f a1 = *(const v4f*)(p + 4);
    v8h h;
    h[0] = (_Float16)a0[0]; h[1] = (_Float16)a0[1];
    h[2] = (_Float16)a0[2]; h[3] = (_Float16)a0[3];
    h[4] = (_Float16)a1[0]; h[5] = (_Float16)a1[1];
    h[6] = (_Float16)a1[2]; h[7] = (_Float16)a1[3];
    return h;
}

// A-fragment (16x32 f16): lane holds row m=(lane&15); halves i:
//   k = 8*(lane>=16) + 16*(i>=8) + (i&7)  -> two 16B LDS loads
__device__ __forceinline__ v16h load_afrag(const _Float16* __restrict__ base,
                                           int lane) {
    const int am  = lane & 15;
    const int akb = (lane >> 4) * 8;
    v8h lo = *(const v8h*)&base[am * AP + akb];
    v8h hi = *(const v8h*)&base[am * AP + akb + 16];
    return __builtin_shufflevector(lo, hi, 0,1,2,3,4,5,6,7,
                                          8,9,10,11,12,13,14,15);
}

// B-fragment (32x16 f16): lane holds col n=(lane&15); halves i:
//   k = 8*(lane>=16) + 16*(i>=8) + (i&7)  -> two 16B LDS loads
__device__ __forceinline__ v16h load_bfrag(const _Float16* __restrict__ base,
                                           int lane) {
    const int bn  = lane & 15;
    const int bkb = (lane >> 4) * 8;
    v8h lo = *(const v8h*)&base[bn * BP + bkb];
    v8h hi = *(const v8h*)&base[bn * BP + bkb + 16];
    return __builtin_shufflevector(lo, hi, 0,1,2,3,4,5,6,7,
                                          8,9,10,11,12,13,14,15);
}

// C = A(MxK, lda) * B(NxK, ldb)^T  into C(MxN, ldc)
// mode 0: none; mode 1: +bias[n]; mode 2: softplus(x + bias[n])
// Requires M % 128 == 0, K % 32 == 0; N ragged vs 64 handled by clamp/guard.
__global__ __launch_bounds__(256)
void gemm_nt_wmma(const float* __restrict__ A, const float* __restrict__ B,
                  float* __restrict__ C, int M, int N, int K,
                  int lda, int ldb, int ldc,
                  const float* __restrict__ bias, int mode)
{
    __shared__ __attribute__((aligned(16))) _Float16 lsA[TM * AP];
    __shared__ __attribute__((aligned(16))) _Float16 lsB[TN * BP];

    const int tid  = threadIdx.x;
    const int lane = tid & 31;
    const int wave = tid >> 5;
    const int wm   = wave & 3;    // wave m-group: rows 32*wm .. 32*wm+31
    const int wn   = wave >> 2;   // wave n-group: cols 32*wn .. 32*wn+31
    const int m0   = blockIdx.y * TM;
    const int n0   = blockIdx.x * TN;

    // A loader: 128 rows x 32 cols, 16 f32 per thread
    const int ar = tid >> 1;               // 0..127
    const int ac = (tid & 1) * 16;         // 0 or 16
    // B loader: 64 rows x 32 cols, 8 f32 per thread (row clamped vs N)
    const int br = tid >> 2;               // 0..63
    const int bc = (tid & 3) * 8;          // 0,8,16,24
    const int bn_clamped = (n0 + br < N) ? (n0 + br) : (N - 1);

    v8f acc[2][2] = {};

    for (int k0 = 0; k0 < K; k0 += KT) {
        __syncthreads();   // protect LDS from previous iteration's readers

        // ---- stage A tile (f32 -> f16) ----
        {
            const float* ga = A + (size_t)(m0 + ar) * lda + (k0 + ac);
            *(v8h*)&lsA[ar * AP + ac]     = cvt_v8(ga);
            *(v8h*)&lsA[ar * AP + ac + 8] = cvt_v8(ga + 8);
        }
        // ---- stage B tile (f32 -> f16), stored [n][k] ----
        {
            const float* gb = B + (size_t)bn_clamped * ldb + (k0 + bc);
            *(v8h*)&lsB[br * BP + bc] = cvt_v8(gb);
        }
        if (k0 + KT < K) {  // prefetch next K tile -> global_prefetch_b8
            __builtin_prefetch(A + (size_t)(m0 + ar) * lda + (k0 + KT + ac), 0, 1);
            __builtin_prefetch(B + (size_t)bn_clamped * ldb + (k0 + KT + bc), 0, 1);
        }
        __syncthreads();

        // ---- 2x2 register-blocked WMMA: 4 matrix ops per wave per K-step ----
        v16h a0 = load_afrag(&lsA[(32 * wm)      * AP], lane);
        v16h a1 = load_afrag(&lsA[(32 * wm + 16) * AP], lane);
        v16h b0 = load_bfrag(&lsB[(32 * wn)      * BP], lane);
        v16h b1 = load_bfrag(&lsB[(32 * wn + 16) * BP], lane);

        acc[0][0] = __builtin_amdgcn_wmma_f32_16x16x32_f16(
            false, a0, false, b0, (short)0, acc[0][0], false, false);
        acc[0][1] = __builtin_amdgcn_wmma_f32_16x16x32_f16(
            false, a0, false, b1, (short)0, acc[0][1], false, false);
        acc[1][0] = __builtin_amdgcn_wmma_f32_16x16x32_f16(
            false, a1, false, b0, (short)0, acc[1][0], false, false);
        acc[1][1] = __builtin_amdgcn_wmma_f32_16x16x32_f16(
            false, a1, false, b1, (short)0, acc[1][1], false, false);
    }

    // ---- epilogue: C/D layout -> m = base + 8*(lane>=16) + r, n = lane&15 ----
    #pragma unroll
    for (int i = 0; i < 2; ++i) {
        const int cm = m0 + 32 * wm + 16 * i + (lane >> 4) * 8;
        #pragma unroll
        for (int j = 0; j < 2; ++j) {
            const int cn = n0 + 32 * wn + 16 * j + (lane & 15);
            if (cn >= N) continue;
            float bv = (mode >= 1) ? bias[cn] : 0.f;
            #pragma unroll
            for (int r = 0; r < 8; ++r) {
                float v = acc[i][j][r] + bv;
                if (mode == 2) v = (v > 20.f) ? v : log1pf(__expf(v));
                C[(size_t)(cm + r) * ldc + cn] = v;
            }
        }
    }
}

// causal depthwise conv1d (k=4) + SiLU.  xc lives in xz cols [0,2048), ld 4096.
__global__ __launch_bounds__(256)
void conv_silu_k(const float* __restrict__ xz, const float* __restrict__ cw,
                 const float* __restrict__ cb, float* __restrict__ out)
{
    int id = blockIdx.x * 256 + threadIdx.x;     // over L*2048
    int e = id & 2047;
    int l = id >> 11;
    float acc = cb[e];
    #pragma unroll
    for (int k = 0; k < 4; ++k) {
        int ls = l + k - 3;
        if (ls >= 0) acc += xz[(size_t)ls * 4096 + e] * cw[e * 4 + k];
    }
    out[(size_t)l * 2048 + e] = acc / (1.f + __expf(-acc));
}

// selective scan fused with D-skip and SiLU gate. One thread per channel e.
__global__ __launch_bounds__(256)
void scan_k(const float* __restrict__ dt, const float* __restrict__ xconv,
            const float* __restrict__ dbc, const float* __restrict__ A_log,
            const float* __restrict__ Dp, const float* __restrict__ xz,
            float* __restrict__ y, int L)
{
    int e = blockIdx.x * 256 + threadIdx.x;      // 0..2047
    float Ae[16], h[16];
    #pragma unroll
    for (int s = 0; s < 16; ++s) { Ae[s] = -__expf(A_log[e * 16 + s]); h[s] = 0.f; }
    const float d = Dp[e];
    for (int l = 0; l < L; ++l) {
        float dtv = dt[(size_t)l * 2048 + e];
        float xv  = xconv[(size_t)l * 2048 + e];
        const float* bc = dbc + (size_t)l * 96;   // B at [64,80), C at [80,96)
        float dx = dtv * xv;
        float yv = 0.f;
        #pragma unroll
        for (int s = 0; s < 16; ++s) {
            h[s] = h[s] * __expf(dtv * Ae[s]) + dx * bc[64 + s];
            yv += h[s] * bc[80 + s];
        }
        float zv = xz[(size_t)l * 4096 + 2048 + e];
        float sz = zv / (1.f + __expf(-zv));
        y[(size_t)l * 2048 + e] = (yv + xv * d) * sz;
    }
}

extern "C" void kernel_launch(void* const* d_in, const int* in_sizes, int n_in,
                              void* d_out, int out_size, void* d_ws, size_t ws_size,
                              hipStream_t stream)
{
    const int L = 2048, DI = 2048;
    (void)in_sizes; (void)n_in; (void)out_size; (void)ws_size;

    // workspace partition (floats)
    float* ws     = (float*)d_ws;
    float* xz     = ws;                                  // L x 4096
    float* xconv  = xz    + (size_t)L * 4096;            // L x 2048
    float* dbc    = xconv + (size_t)L * 2048;            // L x 96
    float* dtb    = dbc   + (size_t)L * 96;              // L x 2048
    float* yb     = dtb   + (size_t)L * 2048;            // L x 2048
    float* cat    = yb    + (size_t)L * 2048;            // L x 2048 (z_l | z_t)

    const float* fusion_w = (const float*)d_in[2];       // (1024, 2048)
    const float* fusion_b = (const float*)d_in[3];       // (1024,)

    dim3 blk(256);
    for (int s = 0; s < 2; ++s) {
        const float* x         = (const float*)d_in[s];           // (L,1024)
        int base = 4 + s * 9;
        const float* in_proj_w = (const float*)d_in[base + 0];    // (4096,1024)
        const float* conv_w    = (const float*)d_in[base + 1];    // (2048,4)
        const float* conv_b    = (const float*)d_in[base + 2];    // (2048,)
        const float* x_proj_w  = (const float*)d_in[base + 3];    // (96,2048)
        const float* dt_proj_w = (const float*)d_in[base + 4];    // (2048,64)
        const float* dt_proj_b = (const float*)d_in[base + 5];    // (2048,)
        const float* A_log     = (const float*)d_in[base + 6];    // (2048,16)
        const float* Dparam    = (const float*)d_in[base + 7];    // (2048,)
        const float* out_proj  = (const float*)d_in[base + 8];    // (1024,2048)

        // xz = x @ in_proj_w^T : M=L, N=4096, K=1024
        gemm_nt_wmma<<<dim3(4096 / TN, L / TM), blk, 0, stream>>>(
            x, in_proj_w, xz, L, 4096, 1024, 1024, 1024, 4096, nullptr, 0);

        // causal conv + SiLU on xc = xz[:, :2048]
        conv_silu_k<<<(L * DI) / 256, blk, 0, stream>>>(xz, conv_w, conv_b, xconv);

        // dbc = xconv @ x_proj_w^T : N=96 (ragged vs 64), K=2048
        gemm_nt_wmma<<<dim3((96 + TN - 1) / TN, L / TM), blk, 0, stream>>>(
            xconv, x_proj_w, dbc, L, 96, 2048, 2048, 2048, 96, nullptr, 0);

        // dt = softplus(dbc[:, :64] @ dt_proj_w^T + dt_proj_b) : N=2048, K=64
        gemm_nt_wmma<<<dim3(2048 / TN, L / TM), blk, 0, stream>>>(
            dbc, dt_proj_w, dtb, L, 2048, 64, 96, 64, 2048, dt_proj_b, 2);

        // selective scan + D skip + SiLU(z) gate
        scan_k<<<DI / 256, blk, 0, stream>>>(dtb, xconv, dbc, A_log, Dparam, xz, yb, L);

        // z_s = y @ out_proj^T into concat buffer (col offset s*1024, ldc=2048)
        gemm_nt_wmma<<<dim3(1024 / TN, L / TM), blk, 0, stream>>>(
            yb, out_proj, cat + s * 1024, L, 1024, 2048, 2048, 2048, 2048, nullptr, 0);
    }

    // out = cat @ fusion_w^T + fusion_b : N=1024, K=2048
    gemm_nt_wmma<<<dim3(1024 / TN, L / TM), blk, 0, stream>>>(
        cat, fusion_w, (float*)d_out, L, 1024, 2048, 2048, 2048, 1024, fusion_b, 1);
}
